// MARN_cell_20504173871344
// MI455X (gfx1250) — matrix operand order repaired and checked by reading, not compiled, not run on hardware
//
#include <hip/hip_runtime.h>

// ---------------------------------------------------------------------------
// MARN fused recurrent cell for gfx1250 (MI455X, wave32, WMMA)
//   T=128, N=1024, P=2, DH=128, D_L=300, D_A=100
// One 16-row WMMA M-tile per workgroup, persistent over all 128 timesteps.
// bf16 WMMA (v_wmma_f32_16x16x32_bf16) with f32 accumulation.
// Round 2: ping-pong h buffers so GRU / LSTM-l / LSTM-a run as independent
// low-register phases (<=6 live C-frags) -> no spills, no VGPR-MSB.
// ---------------------------------------------------------------------------

typedef __attribute__((ext_vector_type(16))) __bf16 v16bf;
typedef __attribute__((ext_vector_type(8)))  float  v8f;

#define T_STEPS 128
#define NBATCH  1024
#define DH      128
#define ROWS    16
#define BLOCK   256

// K-chunk counts (K padded to multiples of 32)
constexpr int KC_GI = 14;  // GRU input: [x_l 300->320] + [x_a 100->128] = 448/32
constexpr int KC_HH = 4;   // 128/32
constexpr int KC_LW = 10;  // 320/32
constexpr int KC_LU = 4;
constexpr int KC_AW = 4;   // 128/32
constexpr int KC_AU = 4;

// bf16 workspace layout (element offsets)
constexpr size_t OFF_GI = 0;
constexpr size_t SZ_GI  = (size_t)24 * KC_GI * 512;   // 24 ntiles (384/16), 512 bf16/frag
constexpr size_t OFF_HH = OFF_GI + SZ_GI;
constexpr size_t SZ_HH  = (size_t)24 * KC_HH * 512;
constexpr size_t OFF_LW = OFF_HH + SZ_HH;
constexpr size_t SZ_LW  = (size_t)32 * KC_LW * 512;   // 32 ntiles (512/16)
constexpr size_t OFF_LU = OFF_LW + SZ_LW;
constexpr size_t SZ_LU  = (size_t)32 * KC_LU * 512;
constexpr size_t OFF_AW = OFF_LU + SZ_LU;
constexpr size_t SZ_AW  = (size_t)32 * KC_AW * 512;
constexpr size_t OFF_AU = OFF_AW + SZ_AW;

__device__ __forceinline__ unsigned short f32_to_bf16(float f) {
  unsigned int u = __float_as_uint(f);
  unsigned int r = (u + 0x7FFFu + ((u >> 16) & 1u)) >> 16;
  return (unsigned short)r;
}
__device__ __forceinline__ float bf16_to_f32(unsigned short b) {
  return __uint_as_float(((unsigned int)b) << 16);
}
__device__ __forceinline__ float fast_sigmoid(float x) { return 1.0f / (1.0f + __expf(-x)); }
__device__ __forceinline__ float fast_tanh(float x)    { return 1.0f - 2.0f / (__expf(2.0f * x) + 1.0f); }

__device__ __forceinline__ v8f wmma_bf16(v16bf a, v16bf b, v8f c) {
  return __builtin_amdgcn_wmma_f32_16x16x32_bf16(false, a, false, b, (short)0, c, false, false);
}

// A-fragment (16x32 bf16) from LDS, ISA 7.12.2 layout:
//  lane L: M = L&15, khalf = L>>4; VGPR r holds K pair k,k+1 with
//  k = (r>=4?16:0) + 2*(r&3) + khalf*8. Pairs are K-consecutive -> 32-bit loads.
__device__ __forceinline__ v16bf load_a_frag(const unsigned short* s, int stride_u32, int kc) {
  union { v16bf v; unsigned int u[8]; } a;
  const unsigned int* p = (const unsigned int*)s;
  unsigned lane = threadIdx.x & 31u;
  unsigned M = lane & 15u, kh = lane >> 4;
  unsigned kb = (unsigned)kc * 32u;
#pragma unroll
  for (int r = 0; r < 8; ++r) {
    unsigned k = ((r >> 2) << 4) + ((r & 3) << 1) + kh * 8u;
    a.u[r] = p[M * stride_u32 + ((kb + k) >> 1)];
  }
  return a.v;
}

// A-fragment from the 2-slot GRU state: row M reads speaker slot sel[M].
// qb -> ushort[2][ROWS][DH]
__device__ __forceinline__ v16bf load_q_frag(const unsigned short* qb, const int* sel, int kc) {
  union { v16bf v; unsigned int u[8]; } a;
  const unsigned int* p = (const unsigned int*)qb;
  unsigned lane = threadIdx.x & 31u;
  unsigned M = lane & 15u, kh = lane >> 4;
  unsigned base = (unsigned)sel[M] * (ROWS * DH / 2) + M * (DH / 2);
  unsigned kb = (unsigned)kc * 32u;
#pragma unroll
  for (int r = 0; r < 8; ++r) {
    unsigned k = ((r >> 2) << 4) + ((r & 3) << 1) + kh * 8u;
    a.u[r] = p[base + ((kb + k) >> 1)];
  }
  return a.v;
}

// B-fragment: pre-swizzled in workspace, 32 lanes x 16 bf16 contiguous per frag.
__device__ __forceinline__ v16bf load_b_frag(const unsigned short* wsb, unsigned fragIdx) {
  unsigned lane = threadIdx.x & 31u;
  return *(const v16bf*)(wsb + ((size_t)fragIdx * 32u + lane) * 16u);
}

// ---------------------------------------------------------------------------
// Weight prep: f32 (K x Ngates) -> bf16 B-fragments, K zero-padded.
// B layout (32x16 K-major tile): lane L: n = L&15; element e holds
// K = (e>>3)*16 + (L>>4)*8 + (e&7).  gru_perm remaps K to [x_l(300)|pad20|x_a(100)|pad28].
// ---------------------------------------------------------------------------
__global__ void prep_weights_kernel(const float* __restrict__ W, unsigned short* __restrict__ dst,
                                    int Ngates, int Ksrc, int KC, int gru_perm) {
  int gid = blockIdx.x * blockDim.x + threadIdx.x;
  int ntiles = Ngates >> 4;
  int total = ntiles * KC * 32;
  if (gid >= total) return;
  int lane = gid & 31;
  int frag = gid >> 5;  // nt*KC + kc
  int kc = frag % KC;
  int nloc = lane & 15;
  int khalf = lane >> 4;
  unsigned short* out = dst + ((size_t)frag * 32 + lane) * 16;
  int n = (frag / KC) * 16 + nloc;
#pragma unroll
  for (int e = 0; e < 16; ++e) {
    int r = e >> 1;
    int k = (e >> 3) * 16 + khalf * 8 + (e & 7);
    int kk = kc * 32 + k;
    int row;
    if (gru_perm) {
      row = (kk < 300) ? kk : ((kk >= 320 && kk < 420) ? (kk - 20) : -1);
    } else {
      row = (kk < Ksrc) ? kk : -1;
    }
    float v = (row >= 0) ? W[(size_t)row * Ngates + n] : 0.0f;
    out[e] = f32_to_bf16(v);
  }
}

// ---------------------------------------------------------------------------
// Fused persistent recurrent kernel: 1 workgroup = 16 batch rows, 8 waves.
// Wave w owns hidden columns [16w, 16w+16) of each 128-wide gate block.
// ---------------------------------------------------------------------------
__global__ __launch_bounds__(BLOCK) void marn_fused_kernel(
    const float* __restrict__ x_l, const float* __restrict__ x_a,
    const int* __restrict__ qmask,
    const float* __restrict__ bl_W, const float* __restrict__ bl_U,
    const float* __restrict__ ba_W, const float* __restrict__ ba_U,
    const float* __restrict__ gru_bih, const float* __restrict__ gru_bhh,
    const float* __restrict__ att_Wq, const float* __restrict__ att_Wk,
    const unsigned short* __restrict__ ws,
    float* __restrict__ out) {
  __shared__ unsigned short xl_s[ROWS][320];     // bf16, x_l padded 300->320
  __shared__ unsigned short xa_s[ROWS][128];     // bf16, x_a padded 100->128
  __shared__ unsigned short hl_s[2][ROWS][DH];   // bf16 h_l, ping-pong by t&1
  __shared__ unsigned short ha_s[2][ROWS][DH];   // bf16 h_a, ping-pong
  __shared__ unsigned short q_s[2][ROWS][DH];    // bf16 GRU speaker states (2 slots)
  __shared__ float cl_s[ROWS][DH];               // c_l (for cross-attn)
  __shared__ float ca_s[ROWS][DH];               // c_a
  __shared__ float wq_s[DH], wk_s[DH];
  __shared__ float qk_s[ROWS];
  __shared__ int sel_s[ROWS];

  const int tid = threadIdx.x;
  const int lane = tid & 31;
  const int wave = tid >> 5;
  const int row0 = blockIdx.x * ROWS;

  for (int i = tid; i < 2 * ROWS * DH; i += BLOCK) {
    ((unsigned short*)hl_s)[i] = 0;
    ((unsigned short*)ha_s)[i] = 0;
    ((unsigned short*)q_s)[i] = 0;
  }
  if (tid < DH) { wq_s[tid] = att_Wq[tid]; wk_s[tid] = att_Wk[tid]; }
  __syncthreads();

  v8f c_l = {};  // persistent LSTM cell state fragments (registers)
  v8f c_a = {};

  const float scale = 0.08838834764831845f;  // 1/sqrt(128)
  const size_t TN = (size_t)T_STEPS * NBATCH;
  float* outH   = out;                  // (T,N,384) = [h_l | h_a | z_l]
  float* outHL  = out + TN * 384;       // (T,N,128)
  float* outHA  = outHL + TN * 128;
  float* outHSP = outHA + TN * 128;
  float* outHLI = outHSP + TN * 128;

  const int colw = wave * 16 + (lane & 15);  // this lane's hidden column
  const int mhalf = lane >> 4;               // C/D layout: M = r + 8*mhalf

  for (int t = 0; t < T_STEPS; ++t) {
    const int p = t & 1;        // read-parity for h ping-pong buffers
    // ---- stage inputs (f32 -> bf16 in LDS, zero-padded) ----
    const float* xlt = x_l + ((size_t)t * NBATCH + row0) * 300;
    const float* xat = x_a + ((size_t)t * NBATCH + row0) * 100;
    for (int i = tid; i < ROWS * 320; i += BLOCK) {
      int rr = i / 320, cc = i - rr * 320;
      xl_s[rr][cc] = f32_to_bf16(cc < 300 ? xlt[(size_t)rr * 300 + cc] : 0.0f);
    }
    for (int i = tid; i < ROWS * 128; i += BLOCK) {
      int rr = i >> 7, cc = i & 127;
      xa_s[rr][cc] = f32_to_bf16(cc < 100 ? xat[(size_t)rr * 100 + cc] : 0.0f);
    }
    if (tid < ROWS) sel_s[tid] = qmask[((size_t)t * NBATCH + row0 + tid) * 2 + 1];
    if (t + 1 < T_STEPS && tid < ROWS) {  // prefetch next step's activations
      __builtin_prefetch(x_l + ((size_t)(t + 1) * NBATCH + row0 + tid) * 300, 0, 3);
      __builtin_prefetch(x_a + ((size_t)(t + 1) * NBATCH + row0 + tid) * 100, 0, 3);
    }
    __syncthreads();

    // ================= GRU phase =================
    {
      v8f gi[3], gh[3];
#pragma unroll
      for (int g = 0; g < 3; ++g) {
        float bi = gru_bih[g * 128 + colw], bh = gru_bhh[g * 128 + colw];
#pragma unroll
        for (int r = 0; r < 8; ++r) { gi[g][r] = bi; gh[g][r] = bh; }
      }
#pragma unroll 2
      for (int kc = 0; kc < KC_GI; ++kc) {
        v16bf a = (kc < KC_LW) ? load_a_frag(&xl_s[0][0], 160, kc)
                               : load_a_frag(&xa_s[0][0], 64, kc - KC_LW);
#pragma unroll
        for (int g = 0; g < 3; ++g)
          gi[g] = wmma_bf16(a, load_b_frag(ws + OFF_GI, (unsigned)((g * 8 + wave) * KC_GI + kc)), gi[g]);
      }
#pragma unroll 2
      for (int kc = 0; kc < KC_HH; ++kc) {
        v16bf a = load_q_frag(&q_s[0][0][0], sel_s, kc);
#pragma unroll
        for (int g = 0; g < 3; ++g)
          gh[g] = wmma_bf16(a, load_b_frag(ws + OFF_HH, (unsigned)((g * 8 + wave) * KC_HH + kc)), gh[g]);
      }
      __syncthreads();  // all waves done reading q_s before owners rewrite it

#pragma unroll
      for (int r = 0; r < 8; ++r) {
        int row = r + mhalf * 8;
        size_t o128 = ((size_t)t * NBATCH + row0 + row) * 128 + colw;
        float rg = fast_sigmoid(gi[0][r] + gh[0][r]);
        float zg = fast_sigmoid(gi[1][r] + gh[1][r]);
        float gg = fast_tanh(gi[2][r] + rg * gh[2][r]);
        int sel = sel_s[row];
        float qs0 = bf16_to_f32(q_s[sel][row][colw]);
        float hs = (1.0f - zg) * gg + zg * qs0;
        outHSP[o128] = hs;
        outHLI[o128] = bf16_to_f32(q_s[1 - sel][row][colw]);  // ql0 (untouched slot)
        q_s[sel][row][colw] = f32_to_bf16(hs);
      }
    }

    // ================= LSTM l phase =================
    {
      v8f sl[4];
#pragma unroll
      for (int g = 0; g < 4; ++g) {
        float b = bl_W[g * 128 + colw] + bl_U[g * 128 + colw];
#pragma unroll
        for (int r = 0; r < 8; ++r) sl[g][r] = b;
      }
#pragma unroll 2
      for (int kc = 0; kc < KC_LW; ++kc) {
        v16bf a = load_a_frag(&xl_s[0][0], 160, kc);
#pragma unroll
        for (int g = 0; g < 4; ++g)
          sl[g] = wmma_bf16(a, load_b_frag(ws + OFF_LW, (unsigned)((g * 8 + wave) * KC_LW + kc)), sl[g]);
      }
#pragma unroll 2
      for (int kc = 0; kc < KC_LU; ++kc) {
        v16bf a = load_a_frag(&hl_s[p][0][0], 64, kc);   // read parity p
#pragma unroll
        for (int g = 0; g < 4; ++g)
          sl[g] = wmma_bf16(a, load_b_frag(ws + OFF_LU, (unsigned)((g * 8 + wave) * KC_LU + kc)), sl[g]);
      }
#pragma unroll
      for (int r = 0; r < 8; ++r) {
        int row = r + mhalf * 8;
        size_t o128 = ((size_t)t * NBATCH + row0 + row) * 128 + colw;
        size_t o384 = ((size_t)t * NBATCH + row0 + row) * 384;
        float cl_new = fast_sigmoid(sl[0][r]) * c_l[r] + fast_sigmoid(sl[1][r]) * fast_tanh(sl[3][r]);
        float hl = fast_tanh(cl_new) * fast_sigmoid(sl[2][r]);
        c_l[r] = cl_new;
        cl_s[row][colw] = cl_new;
        hl_s[p ^ 1][row][colw] = f32_to_bf16(hl);  // write parity p^1 (no hazard)
        outH[o384 + colw] = hl;
        outHL[o128] = hl;
      }
    }

    // ================= LSTM a phase =================
    {
      v8f sa[4];
#pragma unroll
      for (int g = 0; g < 4; ++g) {
        float b = ba_W[g * 128 + colw] + ba_U[g * 128 + colw];
#pragma unroll
        for (int r = 0; r < 8; ++r) sa[g][r] = b;
      }
#pragma unroll 2
      for (int kc = 0; kc < KC_AW; ++kc) {
        v16bf a = load_a_frag(&xa_s[0][0], 64, kc);
#pragma unroll
        for (int g = 0; g < 4; ++g)
          sa[g] = wmma_bf16(a, load_b_frag(ws + OFF_AW, (unsigned)((g * 8 + wave) * KC_AW + kc)), sa[g]);
      }
#pragma unroll 2
      for (int kc = 0; kc < KC_AU; ++kc) {
        v16bf a = load_a_frag(&ha_s[p][0][0], 64, kc);
#pragma unroll
        for (int g = 0; g < 4; ++g)
          sa[g] = wmma_bf16(a, load_b_frag(ws + OFF_AU, (unsigned)((g * 8 + wave) * KC_AU + kc)), sa[g]);
      }
#pragma unroll
      for (int r = 0; r < 8; ++r) {
        int row = r + mhalf * 8;
        size_t o128 = ((size_t)t * NBATCH + row0 + row) * 128 + colw;
        size_t o384 = ((size_t)t * NBATCH + row0 + row) * 384;
        float ca_new = fast_sigmoid(sa[0][r]) * c_a[r] + fast_sigmoid(sa[1][r]) * fast_tanh(sa[3][r]);
        float ha = fast_tanh(ca_new) * fast_sigmoid(sa[2][r]);
        c_a[r] = ca_new;
        ca_s[row][colw] = ca_new;
        ha_s[p ^ 1][row][colw] = f32_to_bf16(ha);
        outH[o384 + 128 + colw] = ha;
        outHA[o128] = ha;
      }
    }
    __syncthreads();  // cl/ca + new h visible; all xl/xa reads done

    // ================= cross attention =================
    if (tid < ROWS) {
      float acc = 0.0f;
      for (int m = 0; m < DH; ++m) acc += ca_s[tid][m] * wq_s[m];
      qk_s[tid] = acc;
    }
    __syncthreads();
    for (int i = tid; i < ROWS * DH; i += BLOCK) {
      int rr = i >> 7, ii = i & 127;
      float aa = cl_s[rr][ii] * qk_s[rr] * scale;
      float num = 0.0f, den = 0.0f;
      for (int m = 0; m < DH; ++m) {
        float e = __expf(aa * wk_s[m]);
        den += e;
        num += e * ca_s[rr][m];
      }
      outH[((size_t)t * NBATCH + row0 + rr) * 384 + 256 + ii] = num / den;
    }
    __syncthreads();  // protect cl/ca/x stages before next step overwrites
  }
}

// ---------------------------------------------------------------------------
extern "C" void kernel_launch(void* const* d_in, const int* in_sizes, int n_in,
                              void* d_out, int out_size, void* d_ws, size_t ws_size,
                              hipStream_t stream) {
  (void)in_sizes; (void)n_in; (void)out_size; (void)ws_size;
  const float* x_l   = (const float*)d_in[1];
  const float* x_a   = (const float*)d_in[2];
  const int*   qmask = (const int*)d_in[3];
  const float* Wl_W  = (const float*)d_in[4];
  const float* bl_W  = (const float*)d_in[5];
  const float* Wl_U  = (const float*)d_in[6];
  const float* bl_U  = (const float*)d_in[7];
  const float* Wa_W  = (const float*)d_in[8];
  const float* ba_W  = (const float*)d_in[9];
  const float* Wa_U  = (const float*)d_in[10];
  const float* ba_U  = (const float*)d_in[11];
  const float* gWih  = (const float*)d_in[12];
  const float* gWhh  = (const float*)d_in[13];
  const float* gbih  = (const float*)d_in[14];
  const float* gbhh  = (const float*)d_in[15];
  const float* aWq   = (const float*)d_in[16];
  const float* aWk   = (const float*)d_in[17];
  unsigned short* ws = (unsigned short*)d_ws;

  auto prep = [&](const float* W, size_t off, int Ng, int Ks, int KC, int perm) {
    int total = (Ng >> 4) * KC * 32;
    prep_weights_kernel<<<(total + 255) / 256, 256, 0, stream>>>(W, ws + off, Ng, Ks, KC, perm);
  };
  prep(gWih, OFF_GI, 384, 400, KC_GI, 1);
  prep(gWhh, OFF_HH, 384, 128, KC_HH, 0);
  prep(Wl_W, OFF_LW, 512, 300, KC_LW, 0);
  prep(Wl_U, OFF_LU, 512, 128, KC_LU, 0);
  prep(Wa_W, OFF_AW, 512, 100, KC_AW, 0);
  prep(Wa_U, OFF_AU, 512, 128, KC_AU, 0);

  marn_fused_kernel<<<NBATCH / ROWS, BLOCK, 0, stream>>>(
      x_l, x_a, qmask, bl_W, bl_U, ba_W, ba_U, gbih, gbhh, aWq, aWk, ws, (float*)d_out);
}
